// SSTv1_20976620273932
// MI455X (gfx1250) — matrix operand order repaired and checked by reading, not compiled, not run on hardware
//
#include <hip/hip_runtime.h>

typedef _Float16 f16;
typedef __attribute__((ext_vector_type(16))) _Float16 v16h;
typedef __attribute__((ext_vector_type(8)))  _Float16 v8h;
typedef __attribute__((ext_vector_type(8)))  float    v8f;

#define DEV __device__ __forceinline__

// ---------------------------------------------------------------- helpers
DEV v8f wmma16(v16h a, v16h b, v8f c) {
  return __builtin_amdgcn_wmma_f32_16x16x32_f16(false, a, false, b, (short)0, c, false, false);
}
DEV v16h mk16(v8h lo, v8h hi) {
  v16h r;
#pragma unroll
  for (int i = 0; i < 8; ++i) { r[i] = lo[i]; r[i + 8] = hi[i]; }
  return r;
}
DEV float wredsum(float v) {
#pragma unroll
  for (int o = 16; o > 0; o >>= 1) v += __shfl_xor(v, o, 32);
  return v;
}
DEV float wredmax(float v) {
#pragma unroll
  for (int o = 16; o > 0; o >>= 1) v = fmaxf(v, __shfl_xor(v, o, 32));
  return v;
}
DEV float gelu_f(float x) { return 0.5f * x * (1.0f + erff(x * 0.70710678118654752f)); }

// ---------------------------------------------------------------- small utility kernels
__global__ void k_fill_h(f16* p, int n) {
  int i = blockIdx.x * 256 + threadIdx.x;
  if (i < n) p[i] = (f16)0.f;
}
__global__ void k_fill_i32(int* p, int val, int n) {
  int i = blockIdx.x * 256 + threadIdx.x;
  if (i < n) p[i] = val;
}
__global__ void k_copy_f32(const float* __restrict__ a, float* __restrict__ b, int n) {
  int i = blockIdx.x * 256 + threadIdx.x;
  if (i < n) b[i] = a[i];
}
__global__ void k_f2h(const float* __restrict__ a, f16* __restrict__ b, int n) {
  int i = blockIdx.x * 256 + threadIdx.x;
  if (i < n) b[i] = (f16)a[i];
}
// conv_w [2][cout128][cin128][3][3] f32 -> wt [2][tap9][cout][cin] f16
__global__ void k_convw(const float* __restrict__ cw, f16* __restrict__ wt) {
  int idx = blockIdx.x;             // 2*9*128
  int cin = threadIdx.x;            // 128
  int i = idx / (9 * 128);
  int rem = idx - i * 9 * 128;
  int tap = rem / 128;
  int cout = rem - tap * 128;
  int ky = tap / 3, kx = tap - ky * 3;
  wt[(((size_t)i * 9 + tap) * 128 + cout) * 128 + cin] =
      (f16)cw[((((size_t)i * 128 + cout) * 128 + cin) * 3 + ky) * 3 + kx];
}
// SSTv1 interleaved sin/cos positional embedding
__global__ void k_pos(const float* __restrict__ ciw, float* __restrict__ pos) {
  int n = blockIdx.x, d = threadIdx.x;     // 128 threads
  int c = d >> 6, r = d & 63, j = r >> 1;
  float xy = ciw[(size_t)n * 2 + c] - 6.0f;
  float arg = xy * powf(10000.0f, -(float)j / 32.0f);
  pos[(size_t)n * 128 + d] = (r & 1) ? cosf(arg) : sinf(arg);
}
// build slot->voxel map (padded-mt layout); -1 = padded slot
__global__ void k_s2v(const int* __restrict__ vx, const int* __restrict__ inds,
                      int* __restrict__ s2v, int n, int mt, int mtp) {
  int j = blockIdx.x * 256 + threadIdx.x;
  if (j >= n) return;
  int ind = inds[j];
  int w = ind / mt;
  int t = ind - w * mt;
  s2v[w * mtp + t] = vx[j];
}
// flat2window gather: wqk = f16(feat+pos), wv = f16(feat), zeros for padded slots
__global__ void k_gather(const float* __restrict__ feat, const float* __restrict__ pos,
                         const int* __restrict__ s2v, f16* __restrict__ wqk,
                         f16* __restrict__ wv) {
  int s = blockIdx.x, d = threadIdx.x;  // 128 threads
  int v = s2v[s];
  float fv = 0.f, pv = 0.f;
  if (v >= 0) { fv = feat[(size_t)v * 128 + d]; pv = pos[(size_t)v * 128 + d]; }
  wqk[(size_t)s * 128 + d] = (f16)(fv + pv);
  wv[(size_t)s * 128 + d] = (f16)fv;
}
// residual + LayerNorm, one wave per voxel row of 128
__global__ __launch_bounds__(32) void k_ln(const float* __restrict__ x,
                                           const float* __restrict__ add,
                                           const float* __restrict__ g,
                                           const float* __restrict__ bta,
                                           float* __restrict__ outf,
                                           f16* __restrict__ out16, int w16) {
  int n = blockIdx.x, l = threadIdx.x;
  const float* xr = x + (size_t)n * 128;
  const float* ar = add + (size_t)n * 128;
  float v[4];
#pragma unroll
  for (int j = 0; j < 4; ++j) v[j] = xr[l + 32 * j] + ar[l + 32 * j];
  float s = v[0] + v[1] + v[2] + v[3];
  s = wredsum(s);
  float mu = s * (1.0f / 128.0f);
  float q = 0.f;
#pragma unroll
  for (int j = 0; j < 4; ++j) { float d = v[j] - mu; q += d * d; }
  q = wredsum(q);
  float inv = 1.0f / sqrtf(q * (1.0f / 128.0f) + 1e-5f);
#pragma unroll
  for (int j = 0; j < 4; ++j) {
    int d = l + 32 * j;
    float yv = (v[j] - mu) * inv * g[d] + bta[d];
    outf[(size_t)n * 128 + d] = yv;
    if (w16) out16[(size_t)n * 128 + d] = (f16)yv;
  }
}
// recover_bev scatter into NHWC f16 canvas
__global__ void k_scatter_bev(const float* __restrict__ feat, const int* __restrict__ coors,
                              f16* __restrict__ canvas) {
  int n = blockIdx.x, d = threadIdx.x;  // 128 threads
  int b = coors[(size_t)n * 4 + 0];
  int y = coors[(size_t)n * 4 + 2];
  int x = coors[(size_t)n * 4 + 3];
  canvas[(((size_t)b * 400 + y) * 400 + x) * 128 + d] = (f16)feat[(size_t)n * 128 + d];
}

// ---------------------------------------------------------------- WMMA GEMM: C = A @ W^T + b
// A f16 [M,K] row-major, W f16 [Nout,K] row-major. Block = 2 waves, tile 32M x 64N.
// EPI 0: f32 out [M,Nout]   1: f16 out   2: GELU->f16 out
// EPI 3: V-transpose store to vT[((w*8+h)*16+d)*KPAD + t]   4: window2flat scatter to attnv f32
template <int EPI>
DEV void epi_store(v8f c, int r0, int col, float bs, int Nout, float* outf, f16* outh,
                   const int* s2v, int MTP, int KPAD) {
  if (EPI == 3) {
    int h = col >> 4, d = col & 15;
    int w = r0 / MTP, t = r0 - w * MTP;
    f16* p = outh + ((size_t)((w << 3) + h) * 16 + d) * KPAD + t;
    v8h pk;
#pragma unroll
    for (int v = 0; v < 8; ++v) pk[v] = (f16)(c[v] + bs);
    *(v8h*)p = pk;
  } else if (EPI == 4) {
#pragma unroll
    for (int v = 0; v < 8; ++v) {
      int vox = s2v[r0 + v];
      if (vox >= 0) outf[(size_t)vox * 128 + col] = c[v] + bs;
    }
  } else {
#pragma unroll
    for (int v = 0; v < 8; ++v) {
      float y = c[v] + bs;
      if (EPI == 2) y = gelu_f(y);
      size_t idx = (size_t)(r0 + v) * Nout + col;
      if (EPI == 0) outf[idx] = y; else outh[idx] = (f16)y;
    }
  }
}

template <int EPI>
__global__ __launch_bounds__(64) void k_gemm(const f16* __restrict__ A,
                                             const f16* __restrict__ W,
                                             const float* __restrict__ bias, int K, int Nout,
                                             float* __restrict__ outf, f16* __restrict__ outh,
                                             const int* __restrict__ s2v, int MTP, int KPAD) {
  int lane = threadIdx.x & 31;
  int wave = threadIdx.x >> 5;
  int rb = blockIdx.y << 5;
  int cb = (blockIdx.x << 6) + (wave << 5);
  int l15 = lane & 15;
  int hi = (lane >> 4) & 1;
  const f16* a0 = A + (size_t)(rb + l15) * K + hi * 8;
  const f16* a1 = a0 + (size_t)16 * K;
  const f16* b0 = W + (size_t)(cb + l15) * K + hi * 8;
  const f16* b1 = b0 + (size_t)16 * K;
  v8f c00 = {}, c01 = {}, c10 = {}, c11 = {};
  for (int k0 = 0; k0 < K; k0 += 32) {
    v16h af0 = mk16(*(const v8h*)(a0 + k0), *(const v8h*)(a0 + k0 + 16));
    v16h af1 = mk16(*(const v8h*)(a1 + k0), *(const v8h*)(a1 + k0 + 16));
    v16h bf0 = mk16(*(const v8h*)(b0 + k0), *(const v8h*)(b0 + k0 + 16));
    v16h bf1 = mk16(*(const v8h*)(b1 + k0), *(const v8h*)(b1 + k0 + 16));
    c00 = wmma16(af0, bf0, c00);
    c01 = wmma16(af0, bf1, c01);
    c10 = wmma16(af1, bf0, c10);
    c11 = wmma16(af1, bf1, c11);
  }
  float bs0 = bias[cb + l15];
  float bs1 = bias[cb + 16 + l15];
  epi_store<EPI>(c00, rb + hi * 8, cb + l15, bs0, Nout, outf, outh, s2v, MTP, KPAD);
  epi_store<EPI>(c01, rb + hi * 8, cb + 16 + l15, bs1, Nout, outf, outh, s2v, MTP, KPAD);
  epi_store<EPI>(c10, rb + 16 + hi * 8, cb + l15, bs0, Nout, outf, outh, s2v, MTP, KPAD);
  epi_store<EPI>(c11, rb + 16 + hi * 8, cb + 16 + l15, bs1, Nout, outf, outh, s2v, MTP, KPAD);
}

// ---------------------------------------------------------------- window attention
// One wave per (window, head, 16-query tile). qk f16 [S,256] (q cols 0..127, k cols 128..255),
// vT f16 [(w*8+h)*16+d][KPAD]. Scores via WMMA (HD=16 zero-padded to K=32), full-row softmax
// in LDS, P@V via WMMA over key chunks of 32.
template <int MTP, int KPAD>
__global__ __launch_bounds__(32) void k_attn(const f16* __restrict__ qk,
                                             const f16* __restrict__ vT,
                                             const int* __restrict__ s2v,
                                             f16* __restrict__ obuf) {
  constexpr int NKT = MTP / 16;
  int lane = threadIdx.x;
  int l15 = lane & 15;
  int hi = (lane >> 4) & 1;
  int bid = blockIdx.x;
  int qt = bid % NKT;
  int h = (bid / NKT) & 7;
  int w = bid / (NKT * 8);

  __shared__ alignas(32) float Srow[16][KPAD];
  __shared__ alignas(32) f16 Prow[16][KPAD];
  __shared__ float maskv[KPAD];

  for (int t = lane; t < KPAD; t += 32)
    maskv[t] = (t < MTP && s2v[w * MTP + t] >= 0) ? 0.0f : -1e30f;

  v8h z8 = {};
  int qrow = w * MTP + qt * 16 + l15;
  v16h qa = mk16(*(const v8h*)(qk + (size_t)qrow * 256 + h * 16 + hi * 8), z8);

  v16h z16 = {};
  for (int kt = 0; kt < NKT; ++kt) {
    int krow = w * MTP + kt * 16 + l15;
    v16h kb = mk16(*(const v8h*)(qk + (size_t)krow * 256 + 128 + h * 16 + hi * 8), z8);
    v8f sc = {};
    sc = wmma16(qa, kb, sc);
#pragma unroll
    for (int v = 0; v < 8; ++v) Srow[v + hi * 8][kt * 16 + l15] = 0.25f * sc[v];
  }
  __syncthreads();

  for (int m = 0; m < 16; ++m) {
    float mx = -1e30f;
    for (int t = lane; t < MTP; t += 32) mx = fmaxf(mx, Srow[m][t] + maskv[t]);
    mx = wredmax(mx);
    float sum = 0.f;
    for (int t = lane; t < MTP; t += 32) {
      float e = expf(Srow[m][t] + maskv[t] - mx);
      Srow[m][t] = e;
      sum += e;
    }
    sum = wredsum(sum);
    float inv = 1.0f / sum;
    for (int t = lane; t < MTP; t += 32) Prow[m][t] = (f16)(Srow[m][t] * inv);
    for (int t = MTP + lane; t < KPAD; t += 32) Prow[m][t] = (f16)0.f;
  }
  __syncthreads();

  v8f o = {};
  const f16* vb = vT + ((size_t)((w << 3) + h) * 16 + l15) * KPAD + hi * 8;
#pragma unroll
  for (int kc = 0; kc < KPAD / 32; ++kc) {
    v16h pa = mk16(*(const v8h*)&Prow[l15][kc * 32 + hi * 8],
                   *(const v8h*)&Prow[l15][kc * 32 + 16 + hi * 8]);
    v16h vf = mk16(*(const v8h*)(vb + kc * 32), *(const v8h*)(vb + kc * 32 + 16));
    o = wmma16(pa, vf, o);
  }
  int orow = w * MTP + qt * 16 + hi * 8;
#pragma unroll
  for (int v = 0; v < 8; ++v)
    obuf[(size_t)(orow + v) * 128 + h * 16 + l15] = (f16)o[v];
}

// ---------------------------------------------------------------- dilated 3x3 conv (NHWC implicit GEMM)
// in f16 NHWC [B,400,400,128]; wt f16 [tap9][cout128][cin128]; dilation 2, pad 2.
// Block 128 = 4 waves, each wave: 16 pixels (x) x 32 couts. FINAL: f32 NCHW out + BN + ReLU.
template <int FINAL>
__global__ __launch_bounds__(128) void k_conv(const f16* __restrict__ in,
                                              const f16* __restrict__ wt,
                                              const float* __restrict__ bg,
                                              const float* __restrict__ bb,
                                              const float* __restrict__ bm,
                                              const float* __restrict__ bv,
                                              f16* __restrict__ outh,
                                              float* __restrict__ outf) {
  int lane = threadIdx.x & 31;
  int wave = threadIdx.x >> 5;
  int l15 = lane & 15;
  int hi = (lane >> 4) & 1;
  int x0 = blockIdx.x << 4;
  int y = blockIdx.y;
  int b = blockIdx.z;
  int cb = wave << 5;
  int px = x0 + l15;
  v8f c0 = {}, c1 = {};
  const f16* w0 = wt + (size_t)(cb + l15) * 128 + hi * 8;
  const f16* w1 = w0 + (size_t)16 * 128;
#pragma unroll
  for (int tap = 0; tap < 9; ++tap) {
    int dy = (tap / 3) * 2 - 2;
    int dx = (tap % 3) * 2 - 2;
    int yy = y + dy, xx = px + dx;
    bool inb = ((unsigned)yy < 400u) && ((unsigned)xx < 400u);
    const f16* ip = in + (((size_t)b * 400 + (inb ? yy : 0)) * 400 + (inb ? xx : 0)) * 128 + hi * 8;
    const f16* wp0 = w0 + (size_t)tap * 128 * 128;
    const f16* wp1 = w1 + (size_t)tap * 128 * 128;
    v8h z8 = {};
#pragma unroll
    for (int kc = 0; kc < 4; ++kc) {
      int k0 = kc * 32;
      v8h alo = inb ? *(const v8h*)(ip + k0) : z8;
      v8h ahi = inb ? *(const v8h*)(ip + k0 + 16) : z8;
      v16h af = mk16(alo, ahi);
      v16h bf0 = mk16(*(const v8h*)(wp0 + k0), *(const v8h*)(wp0 + k0 + 16));
      v16h bf1 = mk16(*(const v8h*)(wp1 + k0), *(const v8h*)(wp1 + k0 + 16));
      c0 = wmma16(af, bf0, c0);
      c1 = wmma16(af, bf1, c1);
    }
  }
#pragma unroll
  for (int f = 0; f < 2; ++f) {
    v8f c = f ? c1 : c0;
    int col = cb + f * 16 + l15;
    float sc = bg[col] / sqrtf(bv[col] + 1e-3f);
    float sh = bb[col] - bm[col] * sc;
#pragma unroll
    for (int v = 0; v < 8; ++v) {
      int pix = x0 + v + hi * 8;
      float yv = fmaxf(c[v] * sc + sh, 0.f);
      if (FINAL)
        outf[(((size_t)b * 128 + col) * 400 + y) * 400 + pix] = yv;
      else
        outh[(((size_t)b * 400 + y) * 400 + pix) * 128 + col] = (f16)yv;
    }
  }
}

// ---------------------------------------------------------------- host driver
extern "C" void kernel_launch(void* const* d_in, const int* in_sizes, int n_in, void* d_out,
                              int out_size, void* d_ws, size_t ws_size, hipStream_t stream) {
  (void)in_sizes; (void)n_in; (void)out_size; (void)ws_size;
  const int N = 60000;
  const int NW0 = 960, MTP0 = 48, KPD0 = 64, S0 = NW0 * MTP0;    // 46080
  const int NW1 = 360, MTP1 = 144, KPD1 = 160, S1 = NW1 * MTP1;  // 51840
  const int LVLN[2] = {24000, 36000};
  const int MTORIG[2] = {36, 144};

  const float* voxel_feat = (const float*)d_in[0];
  const int* coors = (const int*)d_in[1];
  const float* ciw[2] = {(const float*)d_in[2], (const float*)d_in[3]};
  const int* vxin[2][2] = {{(const int*)d_in[4], (const int*)d_in[6]},
                           {(const int*)d_in[8], (const int*)d_in[10]}};
  const int* indsin[2][2] = {{(const int*)d_in[5], (const int*)d_in[7]},
                             {(const int*)d_in[9], (const int*)d_in[11]}};
  const float* ipwf = (const float*)d_in[12];
  const float* ipbf = (const float*)d_in[13];
  const float* owf = (const float*)d_in[14];
  const float* obf = (const float*)d_in[15];
  const float* l1wf = (const float*)d_in[16];
  const float* l1bf = (const float*)d_in[17];
  const float* l2wf = (const float*)d_in[18];
  const float* l2bf = (const float*)d_in[19];
  const float* ln1g = (const float*)d_in[20];
  const float* ln1b = (const float*)d_in[21];
  const float* ln2g = (const float*)d_in[22];
  const float* ln2b = (const float*)d_in[23];
  const float* convwf = (const float*)d_in[24];
  const float* bng = (const float*)d_in[25];
  const float* bnb = (const float*)d_in[26];
  const float* bnm = (const float*)d_in[27];
  const float* bnv = (const float*)d_in[28];

  // workspace carve
  char* p = (char*)d_ws;
  auto carve = [&](size_t bytes) -> char* {
    char* r = p;
    p += (bytes + 255) & ~(size_t)255;
    return r;
  };
  float* feat = (float*)carve((size_t)N * 128 * 4);
  float* attnv = (float*)carve((size_t)N * 128 * 4);
  float* ffno = (float*)carve((size_t)N * 128 * 4);
  f16* ffnh = (f16*)carve((size_t)N * 256 * 2);
  f16* feat16 = (f16*)carve((size_t)N * 128 * 2);
  float* pos[2] = {(float*)carve((size_t)N * 128 * 4), (float*)carve((size_t)N * 128 * 4)};
  int* s2v[2][2];
  s2v[0][0] = (int*)carve((size_t)S0 * 4);
  s2v[0][1] = (int*)carve((size_t)S1 * 4);
  s2v[1][0] = (int*)carve((size_t)S0 * 4);
  s2v[1][1] = (int*)carve((size_t)S1 * 4);
  f16* wqk = (f16*)carve((size_t)S1 * 128 * 2);
  f16* wv = (f16*)carve((size_t)S1 * 128 * 2);
  f16* qkb = (f16*)carve((size_t)S1 * 256 * 2);
  size_t vtmax = (size_t)NW0 * 128 * KPD0;
  if ((size_t)NW1 * 128 * KPD1 > vtmax) vtmax = (size_t)NW1 * 128 * KPD1;
  f16* vT = (f16*)carve(vtmax * 2);
  f16* obuf = (f16*)carve((size_t)S1 * 128 * 2);
  f16* canvas = (f16*)carve((size_t)2 * 400 * 400 * 128 * 2);
  f16* out1 = (f16*)carve((size_t)2 * 400 * 400 * 128 * 2);
  f16* ipw16 = (f16*)carve((size_t)12 * 384 * 128 * 2);
  f16* ow16 = (f16*)carve((size_t)12 * 128 * 128 * 2);
  f16* l1w16 = (f16*)carve((size_t)12 * 256 * 128 * 2);
  f16* l2w16 = (f16*)carve((size_t)12 * 128 * 256 * 2);
  f16* cwt = (f16*)carve((size_t)2 * 9 * 128 * 128 * 2);

  auto blks = [](int n) { return (n + 255) / 256; };

  // ---- one-time prep (re-done each call; deterministic) ----
  k_f2h<<<blks(12 * 384 * 128), 256, 0, stream>>>(ipwf, ipw16, 12 * 384 * 128);
  k_f2h<<<blks(12 * 128 * 128), 256, 0, stream>>>(owf, ow16, 12 * 128 * 128);
  k_f2h<<<blks(12 * 256 * 128), 256, 0, stream>>>(l1wf, l1w16, 12 * 256 * 128);
  k_f2h<<<blks(12 * 128 * 256), 256, 0, stream>>>(l2wf, l2w16, 12 * 128 * 256);
  k_convw<<<2 * 9 * 128, 128, 0, stream>>>(convwf, cwt);
  k_pos<<<N, 128, 0, stream>>>(ciw[0], pos[0]);
  k_pos<<<N, 128, 0, stream>>>(ciw[1], pos[1]);
  for (int s = 0; s < 2; ++s)
    for (int lvl = 0; lvl < 2; ++lvl) {
      int S = lvl ? S1 : S0;
      int MTP = lvl ? MTP1 : MTP0;
      k_fill_i32<<<blks(S), 256, 0, stream>>>(s2v[s][lvl], -1, S);
      k_s2v<<<blks(LVLN[lvl]), 256, 0, stream>>>(vxin[s][lvl], indsin[s][lvl], s2v[s][lvl],
                                                 LVLN[lvl], MTORIG[lvl], MTP);
    }
  k_copy_f32<<<blks(N * 128), 256, 0, stream>>>(voxel_feat, feat, N * 128);

  // ---- 12 encoder layers ----
  for (int li = 0; li < 12; ++li) {
    int s = li & 1;
    const f16* Wqk = ipw16 + (size_t)li * 384 * 128;
    const f16* Wv = Wqk + (size_t)256 * 128;
    const float* Bqk = ipbf + (size_t)li * 384;
    const float* Bv = Bqk + 256;
    for (int lvl = 0; lvl < 2; ++lvl) {
      int S = lvl ? S1 : S0;
      int MTP = lvl ? MTP1 : MTP0;
      int KPD = lvl ? KPD1 : KPD0;
      int* sv = s2v[s][lvl];
      k_gather<<<S, 128, 0, stream>>>(feat, pos[s], sv, wqk, wv);
      // q,k projection -> qkb f16 [S,256]
      k_gemm<1><<<dim3(256 / 64, S / 32), 64, 0, stream>>>(wqk, Wqk, Bqk, 128, 256, nullptr,
                                                           qkb, nullptr, 0, 0);
      // v projection -> vT (d-major transposed)
      k_gemm<3><<<dim3(128 / 64, S / 32), 64, 0, stream>>>(wv, Wv, Bv, 128, 128, nullptr, vT,
                                                           nullptr, MTP, KPD);
      if (lvl == 0)
        k_attn<48, 64><<<NW0 * 8 * (MTP0 / 16), 32, 0, stream>>>(qkb, vT, sv, obuf);
      else
        k_attn<144, 160><<<NW1 * 8 * (MTP1 / 16), 32, 0, stream>>>(qkb, vT, sv, obuf);
      // out projection + window2flat scatter into attnv (levels partition all voxels)
      k_gemm<4><<<dim3(128 / 64, S / 32), 64, 0, stream>>>(
          obuf, ow16 + (size_t)li * 128 * 128, obf + (size_t)li * 128, 128, 128, attnv,
          nullptr, sv, 0, 0);
    }
    k_ln<<<N, 32, 0, stream>>>(feat, attnv, ln1g + (size_t)li * 128, ln1b + (size_t)li * 128,
                               feat, feat16, 1);
    k_gemm<2><<<dim3(256 / 64, N / 32), 64, 0, stream>>>(
        feat16, l1w16 + (size_t)li * 256 * 128, l1bf + (size_t)li * 256, 128, 256, nullptr,
        ffnh, nullptr, 0, 0);
    k_gemm<0><<<dim3(128 / 64, N / 32), 64, 0, stream>>>(
        ffnh, l2w16 + (size_t)li * 128 * 256, l2bf + (size_t)li * 128, 256, 128, ffno, nullptr,
        nullptr, 0, 0);
    k_ln<<<N, 32, 0, stream>>>(feat, ffno, ln2g + (size_t)li * 128, ln2b + (size_t)li * 128,
                               feat, feat16, 0);
  }

  // ---- BEV scatter + dilated convs ----
  int cvn = 2 * 400 * 400 * 128;
  k_fill_h<<<blks(cvn), 256, 0, stream>>>(canvas, cvn);
  k_scatter_bev<<<N, 128, 0, stream>>>(feat, coors, canvas);
  k_conv<0><<<dim3(25, 400, 2), 128, 0, stream>>>(canvas, cwt, bng, bnb, bnm, bnv, out1,
                                                  nullptr);
  k_conv<1><<<dim3(25, 400, 2), 128, 0, stream>>>(out1, cwt + (size_t)9 * 128 * 128, bng + 128,
                                                  bnb + 128, bnm + 128, bnv + 128, nullptr,
                                                  (float*)d_out);
}
